// UDASoftLabel_MultiScale_V2_44547400794403
// MI455X (gfx1250) — matrix operand order repaired and checked by reading, not compiled, not run on hardware
//
#include <hip/hip_runtime.h>

// ---------------------------------------------------------------------------
// Types for CDNA5 WMMA (wave32, gfx1250)
// ---------------------------------------------------------------------------
typedef __attribute__((ext_vector_type(8)))  __bf16 v8bf;
typedef __attribute__((ext_vector_type(16))) __bf16 v16bf;
typedef __attribute__((ext_vector_type(8)))  float  v8f;

#define NQ_ROWS   4096             // 64 * 8 * 8
#define NS_ROWS   40960            // 128*16*16 + 128*8*8
#define CDIM      512
#define MBLK      4                // 4 x 16 = 64 query rows per workgroup
#define MROWS     (MBLK * 16)
#define NBLK      2                // 2 x 16 = 32 sim columns per wave-iter
#define N_MACRO   (NS_ROWS / (NBLK * 16))   // 1280
#define TOPK_OUT  15
#define TOPK_IN   4

// ---------------------------------------------------------------------------
// Zero the scalar output (graph-capture safe, deterministic)
// ---------------------------------------------------------------------------
__global__ void zero_out_kernel(float* out) {
    if (threadIdx.x == 0 && blockIdx.x == 0) out[0] = 0.0f;
}

// ---------------------------------------------------------------------------
// Pool (PxP average, P=2 or 4) one output row, L2-normalize it, emit bf16.
// One workgroup (256 threads) per output row; each thread handles 2 channels.
// ---------------------------------------------------------------------------
template <int P>
__global__ __launch_bounds__(256)
void pool_norm_kernel(const float* __restrict__ in,
                      __bf16* __restrict__ out,
                      int oHW, int oW, int rowOffset) {
    const int row = blockIdx.x;
    const int b   = row / oHW;
    const int rem = row % oHW;
    const int h   = rem / oW;
    const int w   = rem % oW;

    __shared__ float red[256];

    const float inv = 1.0f / (float)(P * P);
    float vals[2];
    float ss = 0.0f;

#pragma unroll
    for (int i = 0; i < 2; ++i) {
        const int c = threadIdx.x + i * 256;
        const float* base =
            in + (((size_t)b * CDIM + c) * 32 + h * P) * 32 + w * P;
        float s = 0.0f;
#pragma unroll
        for (int y = 0; y < P; ++y)
#pragma unroll
            for (int x = 0; x < P; ++x)
                s += base[y * 32 + x];
        s *= inv;
        vals[i] = s;
        ss += s * s;
    }

    red[threadIdx.x] = ss;
    __syncthreads();
#pragma unroll
    for (int off = 128; off > 0; off >>= 1) {
        if (threadIdx.x < off) red[threadIdx.x] += red[threadIdx.x + off];
        __syncthreads();
    }
    const float rinv = rsqrtf(red[0]);

    __bf16* orow = out + (size_t)(rowOffset + row) * CDIM;
#pragma unroll
    for (int i = 0; i < 2; ++i) {
        const int c = threadIdx.x + i * 256;
        orow[c] = (__bf16)(vals[i] * rinv);
    }
}

// ---------------------------------------------------------------------------
// Fused cosine-sim GEMM (bf16 WMMA, register-blocked 64x32) + streaming
// per-row top-15 + loss.
// Grid: 64 blocks (64 query rows each), 256 threads (8 waves).
// Each wave-iteration: 8 WMMAs per K-step (4 M-blocks x 2 N-blocks) sharing
// one B fragment stream -> 64 FLOP per B byte from L2, 1 ds + 0.5 vmem
// load per WMMA.
// ---------------------------------------------------------------------------
__global__ __launch_bounds__(256)
void sim_topk_kernel(const __bf16* __restrict__ qn,
                     const __bf16* __restrict__ sn,
                     float* __restrict__ out) {
    __shared__ __bf16 Ash[MROWS * CDIM];     // 64 KB: 64 normalized q rows
    __shared__ float  tileSh[8][16 * 32];    // 16 KB: per-wave 16x32 spill
    __shared__ float  topSh[8][MROWS][16];   // 32 KB: per-wave per-row top-15
    __shared__ float  fin[MROWS][16];        //  4 KB: final merged top-15
    __shared__ float  lred[MROWS];

    const int tid     = threadIdx.x;
    const int wid     = tid >> 5;
    const int lane    = tid & 31;
    const int ln      = lane & 15;
    const int rowBase = blockIdx.x * MROWS;

    // Stage the 64-row A panel into LDS (65536 bytes, 16x uint4 per thread).
    {
        const uint4* src = (const uint4*)(qn + (size_t)rowBase * CDIM);
        uint4* dst = (uint4*)Ash;
#pragma unroll
        for (int i = 0; i < 16; ++i) dst[tid + i * 256] = src[tid + i * 256];
    }
    // Init per-wave top lists to -inf.
    for (int i = tid; i < 8 * MROWS * 16; i += 256)
        ((float*)topSh)[i] = -1e30f;
    __syncthreads();

    // A fragment addressing, contiguous k order (same order as B):
    //   lanes 0-15 : K = 0..15 contiguous, lanes 16-31 : K = 16..31
    const int ka = (lane < 16) ? 0 : 16;
    // B fragment addressing, contiguous k order (same order as A):
    //   lanes 0-15 : K = 0..15 contiguous, lanes 16-31 : K = 16..31
    const int kb = (lane < 16) ? 0 : 16;

    for (int nt = wid; nt < N_MACRO; nt += 8) {
        const int nbase = nt * (NBLK * 16);
        const __bf16* Brow0 = sn + (size_t)(nbase + ln) * CDIM;
        const __bf16* Brow1 = sn + (size_t)(nbase + 16 + ln) * CDIM;

        // Prefetch this wave's next B macro-panel into cache.
        if (nt + 8 < N_MACRO) {
            __builtin_prefetch(sn + (size_t)(nbase + 256 + ln) * CDIM, 0, 3);
            __builtin_prefetch(sn + (size_t)(nbase + 272 + ln) * CDIM, 0, 3);
        }

        v8f acc[MBLK][NBLK];
#pragma unroll
        for (int m = 0; m < MBLK; ++m)
#pragma unroll
            for (int n = 0; n < NBLK; ++n) acc[m][n] = (v8f)0.0f;

#pragma unroll 4
        for (int ks = 0; ks < CDIM / 32; ++ks) {
            const v16bf b0 = *(const v16bf*)(Brow0 + ks * 32 + kb);
            const v16bf b1 = *(const v16bf*)(Brow1 + ks * 32 + kb);
#pragma unroll
            for (int m = 0; m < MBLK; ++m) {
                const __bf16* Ar = Ash + (size_t)(m * 16 + ln) * CDIM;
                const v8bf alo = *(const v8bf*)(Ar + ks * 32 + ka);
                const v8bf ahi = *(const v8bf*)(Ar + ks * 32 + ka + 8);
                const v16bf a = __builtin_shufflevector(alo, ahi,
                    0, 1, 2, 3, 4, 5, 6, 7, 8, 9, 10, 11, 12, 13, 14, 15);
                acc[m][0] = __builtin_amdgcn_wmma_f32_16x16x32_bf16(
                    false, a, false, b0, (short)0, acc[m][0], false, false);
                acc[m][1] = __builtin_amdgcn_wmma_f32_16x16x32_bf16(
                    false, a, false, b1, (short)0, acc[m][1], false, false);
            }
        }

        // Per M-block: spill 16x32 tile to LDS, then lanes 0-15 merge their
        // row's 32 candidates into the sorted per-wave top-15 list.
#pragma unroll
        for (int m = 0; m < MBLK; ++m) {
#pragma unroll
            for (int j = 0; j < 8; ++j) {
                const int r = (lane < 16) ? j : (j + 8);
                tileSh[wid][r * 32 + ln]      = acc[m][0][j];
                tileSh[wid][r * 32 + 16 + ln] = acc[m][1][j];
            }
            asm volatile("s_wait_dscnt 0" ::: "memory");

            if (lane < 16) {
                float* list = &topSh[wid][m * 16 + lane][0];
                const float* trow = &tileSh[wid][lane * 32];
                for (int i = 0; i < 32; ++i) {
                    const float v = trow[i];
                    if (v > list[TOPK_OUT - 1]) {
                        int p = TOPK_OUT - 1;
                        while (p > 0 && list[p - 1] < v) {
                            list[p] = list[p - 1];
                            --p;
                        }
                        list[p] = v;
                    }
                }
            }
            asm volatile("s_wait_dscnt 0" ::: "memory");
        }
    }

    __syncthreads();

    // Final merge: thread r (< 64) merges the 8 wave-local lists of its row,
    // then loss_row = logsumexp(top15) - mean(top4).
    if (tid < MROWS) {
        float* f = &fin[tid][0];
        for (int k = 0; k < 16; ++k) f[k] = -1e30f;
        for (int wv = 0; wv < 8; ++wv) {
            const float* src = &topSh[wv][tid][0];
            for (int k = 0; k < TOPK_OUT; ++k) {
                const float v = src[k];
                if (v > f[TOPK_OUT - 1]) {
                    int p = TOPK_OUT - 1;
                    while (p > 0 && f[p - 1] < v) {
                        f[p] = f[p - 1];
                        --p;
                    }
                    f[p] = v;
                }
            }
        }
        const float m = f[0];
        float s = 0.0f;
        for (int k = 0; k < TOPK_OUT; ++k) s += __expf(f[k] - m);
        const float lse   = m + __logf(s);
        const float mean4 = (f[0] + f[1] + f[2] + f[3]) * (1.0f / TOPK_IN);
        lred[tid] = (lse - mean4) * (1.0f / (float)NQ_ROWS);
    }
    __syncthreads();
    if (tid == 0) {
        float acc = 0.0f;
        for (int k = 0; k < MROWS; ++k) acc += lred[k];
        atomicAdd(out, acc);
    }
}

// ---------------------------------------------------------------------------
// Launcher
// ---------------------------------------------------------------------------
extern "C" void kernel_launch(void* const* d_in, const int* in_sizes, int n_in,
                              void* d_out, int out_size, void* d_ws,
                              size_t ws_size, hipStream_t stream) {
    (void)in_sizes; (void)n_in; (void)out_size; (void)ws_size;
    const float* q = (const float*)d_in[0];   // [64, 512, 32, 32]
    const float* S = (const float*)d_in[1];   // [128, 512, 32, 32]
    float* out = (float*)d_out;

    char* ws = (char*)d_ws;
    __bf16* qn = (__bf16*)ws;                                   // 4 MB
    __bf16* sn = (__bf16*)(ws + (size_t)NQ_ROWS * CDIM * 2);    // 42 MB

    zero_out_kernel<<<1, 64, 0, stream>>>(out);

    // qf = normalize(avgpool x2 (q))  -> 4096 rows
    pool_norm_kernel<4><<<NQ_ROWS, 256, 0, stream>>>(q, qn, 64, 8, 0);
    // s1 = normalize(avgpool x1 (S))  -> rows [0, 32768)
    pool_norm_kernel<2><<<32768, 256, 0, stream>>>(S, sn, 256, 16, 0);
    // s_src = normalize(avgpool x2 (S)) -> rows [32768, 40960)
    pool_norm_kernel<4><<<8192, 256, 0, stream>>>(S, sn, 64, 8, 32768);

    // Fused GEMM + top-k + loss
    sim_topk_kernel<<<NQ_ROWS / MROWS, 256, 0, stream>>>(qn, sn, out);
}
